// QuantizedLinear_11330123726992
// MI455X (gfx1250) — compile-verified, matrix operand order checked by reading, not run on hardware
//
#include <hip/hip_runtime.h>

typedef _Float16 v16h __attribute__((ext_vector_type(16)));
typedef _Float16 v8h  __attribute__((ext_vector_type(8)));
typedef _Float16 v4h  __attribute__((ext_vector_type(4)));
typedef float    v8f  __attribute__((ext_vector_type(8)));

#define TILE_M 128
#define TILE_N 128
#define TILE_K 32
#define LDS_STRIDE 40   // 32 halves + 8 pad halves -> 80 B row, 16B-aligned chunks

// Byte offset of a shared-memory object inside LDS (addrspace(3) ptr == LDS offset).
__device__ __forceinline__ unsigned lds_off(const void* p) {
    return (unsigned)(size_t)(const __attribute__((address_space(3))) void*)p;
}

// ---------------- stage 0: scale init + abs-max reductions ----------------

__global__ void init_ws_kernel(unsigned* ws) {
    if (threadIdx.x < 2) ws[threadIdx.x] = 0u;   // bits of +0.0f
}

__global__ void absmax_kernel(const float* __restrict__ in, unsigned long long n,
                              unsigned* out) {
    __shared__ float red[256];
    float m = 0.0f;
    for (unsigned long long i = (unsigned long long)blockIdx.x * blockDim.x + threadIdx.x;
         i < n; i += (unsigned long long)gridDim.x * blockDim.x)
        m = fmaxf(m, fabsf(in[i]));
    red[threadIdx.x] = m;
    __syncthreads();
    for (int s = 128; s > 0; s >>= 1) {
        if ((int)threadIdx.x < s)
            red[threadIdx.x] = fmaxf(red[threadIdx.x], red[threadIdx.x + s]);
        __syncthreads();
    }
    if (threadIdx.x == 0)
        atomicMax(out, __float_as_uint(red[0]));   // valid order for floats >= 0
}

// ---------------- stage 1 (big-ws path): pack fp32 -> fp16 ----------------

__global__ void pack_x_kernel(const float* __restrict__ x, _Float16* __restrict__ xh,
                              unsigned long long n4) {
    unsigned long long i = (unsigned long long)blockIdx.x * blockDim.x + threadIdx.x;
    if (i >= n4) return;
    float4 v = ((const float4*)x)[i];
    v4h h;
    h[0] = (_Float16)v.x; h[1] = (_Float16)v.y;
    h[2] = (_Float16)v.z; h[3] = (_Float16)v.w;
    ((v4h*)xh)[i] = h;
}

__global__ void pack_w_kernel(const float* __restrict__ w, _Float16* __restrict__ wq,
                              const float* __restrict__ scales, unsigned long long n4) {
    unsigned long long i = (unsigned long long)blockIdx.x * blockDim.x + threadIdx.x;
    if (i >= n4) return;
    const float inv_wscale = 127.0f / scales[0];
    float4 v = ((const float4*)w)[i];
    v4h h;
    h[0] = (_Float16)fminf(127.f, fmaxf(-128.f, rintf(v.x * inv_wscale)));
    h[1] = (_Float16)fminf(127.f, fmaxf(-128.f, rintf(v.y * inv_wscale)));
    h[2] = (_Float16)fminf(127.f, fmaxf(-128.f, rintf(v.z * inv_wscale)));
    h[3] = (_Float16)fminf(127.f, fmaxf(-128.f, rintf(v.w * inv_wscale)));
    ((v4h*)wq)[i] = h;
}

// ---------------- fused path: fp32 in, convert/quant while staging ----------------

__global__ __launch_bounds__(256)
void wmma_qlinear_fused_kernel(const float* __restrict__ X,   // [M,K]
                               const float* __restrict__ W,   // [N,K]
                               const float* __restrict__ Bv,  // [N]
                               const float* __restrict__ scales,
                               float* __restrict__ out,       // [M,N]
                               int M, int N, int K) {
    __shared__ __align__(16) _Float16 As[2][TILE_M * LDS_STRIDE];
    __shared__ __align__(16) _Float16 Bs[2][TILE_N * LDS_STRIDE];

    const int tid  = threadIdx.x;
    const int lane = tid & 31;
    const int wave = tid >> 5;
    const int wm   = wave >> 1;
    const int wn   = wave & 1;
    const int m0   = blockIdx.y * TILE_M;
    const int n0   = blockIdx.x * TILE_N;

    const float wmax       = scales[0];
    const float wscale     = wmax * (1.0f / 127.0f);
    const float inv_wscale = 127.0f / wmax;
    const float bmax       = scales[1];
    const float bscale     = bmax * (1.0f / 127.0f);
    const float inv_bscale = 127.0f / bmax;

    v8f acc[2][4];
#pragma unroll
    for (int i = 0; i < 2; ++i)
#pragma unroll
        for (int j = 0; j < 4; ++j)
#pragma unroll
            for (int e = 0; e < 8; ++e) acc[i][j][e] = 0.0f;

    const int half_sel = lane >> 4;
    const int mrow     = lane & 15;

    float4 xr[4], wr[4];

    auto load_regs = [&](int kt) {
#pragma unroll
        for (int i = 0; i < 4; ++i) {
            const int idx = tid + i * 256;
            const int row = idx >> 3;
            const int c4  = (idx & 7) << 2;
            xr[i] = *(const float4*)(X + (size_t)(m0 + row) * K + kt + c4);
            wr[i] = *(const float4*)(W + (size_t)(n0 + row) * K + kt + c4);
        }
    };

    auto store_lds = [&](int p) {
#pragma unroll
        for (int i = 0; i < 4; ++i) {
            const int idx = tid + i * 256;
            const int row = idx >> 3;
            const int c4  = (idx & 7) << 2;
            v4h xh;
            xh[0] = (_Float16)xr[i].x; xh[1] = (_Float16)xr[i].y;
            xh[2] = (_Float16)xr[i].z; xh[3] = (_Float16)xr[i].w;
            *(v4h*)(&As[p][row * LDS_STRIDE + c4]) = xh;
            v4h wq;
            wq[0] = (_Float16)fminf(127.f, fmaxf(-128.f, rintf(wr[i].x * inv_wscale)));
            wq[1] = (_Float16)fminf(127.f, fmaxf(-128.f, rintf(wr[i].y * inv_wscale)));
            wq[2] = (_Float16)fminf(127.f, fmaxf(-128.f, rintf(wr[i].z * inv_wscale)));
            wq[3] = (_Float16)fminf(127.f, fmaxf(-128.f, rintf(wr[i].w * inv_wscale)));
            *(v4h*)(&Bs[p][row * LDS_STRIDE + c4]) = wq;
        }
    };

    auto compute = [&](int p) {
        v16h afrag[2], bfrag[4];
#pragma unroll
        for (int i = 0; i < 2; ++i) {
            const _Float16* q = &As[p][(wm * 32 + i * 16 + mrow) * LDS_STRIDE + half_sel * 8];
            v8h lo = *(const v8h*)(q);
            v8h hi = *(const v8h*)(q + 16);
            afrag[i] = __builtin_shufflevector(lo, hi, 0, 1, 2, 3, 4, 5, 6, 7,
                                               8, 9, 10, 11, 12, 13, 14, 15);
        }
#pragma unroll
        for (int j = 0; j < 4; ++j) {
            const _Float16* q = &Bs[p][(wn * 64 + j * 16 + mrow) * LDS_STRIDE + half_sel * 8];
            v8h lo = *(const v8h*)(q);
            v8h hi = *(const v8h*)(q + 16);
            bfrag[j] = __builtin_shufflevector(lo, hi, 0, 1, 2, 3, 4, 5, 6, 7,
                                               8, 9, 10, 11, 12, 13, 14, 15);
        }
#pragma unroll
        for (int i = 0; i < 2; ++i)
#pragma unroll
            for (int j = 0; j < 4; ++j)
                acc[i][j] = __builtin_amdgcn_wmma_f32_16x16x32_f16(
                    false, afrag[i], false, bfrag[j], (short)0, acc[i][j], false, false);
    };

    load_regs(0);
    store_lds(0);
    int p = 0;
    for (int kt = 0; kt < K; kt += TILE_K) {
        __syncthreads();
        const bool has_next = (kt + TILE_K) < K;
        if (has_next) load_regs(kt + TILE_K);   // issue early; latency hidden by WMMA
        compute(p);
        if (has_next) store_lds(p ^ 1);
        p ^= 1;
    }

    const int mbase = m0 + wm * 32;
    const int nbase = n0 + wn * 64;
#pragma unroll
    for (int j = 0; j < 4; ++j) {
        const int ng = nbase + j * 16 + mrow;
        const float bq =
            fminf(127.f, fmaxf(-128.f, rintf(Bv[ng] * inv_bscale))) * bscale;
#pragma unroll
        for (int i = 0; i < 2; ++i)
#pragma unroll
            for (int r = 0; r < 8; ++r) {
                const int mg = mbase + i * 16 + r + half_sel * 8;
                out[(size_t)mg * N + ng] = acc[i][j][r] * wscale + bq;
            }
    }
}

// ---------------- big-ws path: fp16 GEMM with async global->LDS staging ----------------

__global__ __launch_bounds__(256)
void wmma_gemm_f16_async_kernel(const _Float16* __restrict__ Xh,  // [M,K] fp16
                                const _Float16* __restrict__ Wq,  // [N,K] fp16 (int8 levels)
                                const float* __restrict__ Bv,
                                const float* __restrict__ scales,
                                float* __restrict__ out,
                                int M, int N, int K) {
    __shared__ __align__(16) _Float16 As[2][TILE_M * LDS_STRIDE];
    __shared__ __align__(16) _Float16 Bs[2][TILE_N * LDS_STRIDE];

    const int tid  = threadIdx.x;
    const int lane = tid & 31;
    const int wave = tid >> 5;
    const int wm   = wave >> 1;
    const int wn   = wave & 1;
    const int m0   = blockIdx.y * TILE_M;
    const int n0   = blockIdx.x * TILE_N;

    const float wscale = scales[0] * (1.0f / 127.0f);
    const float bmax   = scales[1];
    const float bscale = bmax * (1.0f / 127.0f);
    const float inv_bscale = 127.0f / bmax;

    v8f acc[2][4];
#pragma unroll
    for (int i = 0; i < 2; ++i)
#pragma unroll
        for (int j = 0; j < 4; ++j)
#pragma unroll
            for (int e = 0; e < 8; ++e) acc[i][j][e] = 0.0f;

    const int half_sel = lane >> 4;
    const int mrow     = lane & 15;

    const unsigned as_base = lds_off(&As[0][0]);
    const unsigned bs_base = lds_off(&Bs[0][0]);
    const unsigned buf_bytes = TILE_M * LDS_STRIDE * 2;   // 10240 B per buffer

    // Each thread DMAs 2 x 16B chunks per tile per matrix, straight into LDS.
    auto stage_async = [&](int p, int kt) {
        const _Float16* xb = Xh + (size_t)m0 * K + kt;   // uniform -> SGPR pair
        const _Float16* wb = Wq + (size_t)n0 * K + kt;
#pragma unroll
        for (int i = 0; i < 2; ++i) {
            const int idx = tid + i * 256;     // 0..511 chunk id
            const int row = idx >> 2;          // 0..127
            const int cb  = (idx & 3) << 4;    // 0,16,32,48 byte column
            const unsigned voff = (unsigned)(row * (K * 2) + cb);
            const unsigned ldsx = as_base + (unsigned)p * buf_bytes + (unsigned)(row * 80 + cb);
            const unsigned ldsw = bs_base + (unsigned)p * buf_bytes + (unsigned)(row * 80 + cb);
            asm volatile("global_load_async_to_lds_b128 %0, %1, %2 offset:0"
                         :: "v"(ldsx), "v"(voff), "s"(xb) : "memory");
            asm volatile("global_load_async_to_lds_b128 %0, %1, %2 offset:0"
                         :: "v"(ldsw), "v"(voff), "s"(wb) : "memory");
        }
    };

    auto compute = [&](int p) {
        v16h afrag[2], bfrag[4];
#pragma unroll
        for (int i = 0; i < 2; ++i) {
            const _Float16* q = &As[p][(wm * 32 + i * 16 + mrow) * LDS_STRIDE + half_sel * 8];
            v8h lo = *(const v8h*)(q);
            v8h hi = *(const v8h*)(q + 16);
            afrag[i] = __builtin_shufflevector(lo, hi, 0, 1, 2, 3, 4, 5, 6, 7,
                                               8, 9, 10, 11, 12, 13, 14, 15);
        }
#pragma unroll
        for (int j = 0; j < 4; ++j) {
            const _Float16* q = &Bs[p][(wn * 64 + j * 16 + mrow) * LDS_STRIDE + half_sel * 8];
            v8h lo = *(const v8h*)(q);
            v8h hi = *(const v8h*)(q + 16);
            bfrag[j] = __builtin_shufflevector(lo, hi, 0, 1, 2, 3, 4, 5, 6, 7,
                                               8, 9, 10, 11, 12, 13, 14, 15);
        }
#pragma unroll
        for (int i = 0; i < 2; ++i)
#pragma unroll
            for (int j = 0; j < 4; ++j)
                acc[i][j] = __builtin_amdgcn_wmma_f32_16x16x32_f16(
                    false, afrag[i], false, bfrag[j], (short)0, acc[i][j], false, false);
    };

    stage_async(0, 0);
    int p = 0;
    for (int kt = 0; kt < K; kt += TILE_K) {
        asm volatile("s_wait_asynccnt 0x0" ::: "memory");  // own DMAs into LDS done
        __syncthreads();                                   // everyone's DMAs visible
        if (kt + TILE_K < K) stage_async(p ^ 1, kt + TILE_K);  // overlap with WMMA
        compute(p);
        p ^= 1;
    }

    const int mbase = m0 + wm * 32;
    const int nbase = n0 + wn * 64;
#pragma unroll
    for (int j = 0; j < 4; ++j) {
        const int ng = nbase + j * 16 + mrow;
        const float bq =
            fminf(127.f, fmaxf(-128.f, rintf(Bv[ng] * inv_bscale))) * bscale;
#pragma unroll
        for (int i = 0; i < 2; ++i)
#pragma unroll
            for (int r = 0; r < 8; ++r) {
                const int mg = mbase + i * 16 + r + half_sel * 8;
                out[(size_t)mg * N + ng] = acc[i][j][r] * wscale + bq;
            }
    }
}

// ---------------- host ----------------

extern "C" void kernel_launch(void* const* d_in, const int* in_sizes, int n_in,
                              void* d_out, int out_size, void* d_ws, size_t ws_size,
                              hipStream_t stream) {
    const float* x  = (const float*)d_in[0];   // [B,S,IN] fp32
    const float* W  = (const float*)d_in[1];   // [OUT,IN] fp32
    const float* b  = (const float*)d_in[2];   // [OUT]    fp32
    float* out      = (float*)d_out;
    unsigned* ws    = (unsigned*)d_ws;

    const int OUT = in_sizes[2];
    const int IN  = in_sizes[1] / OUT;
    const int M   = in_sizes[0] / IN;

    init_ws_kernel<<<1, 32, 0, stream>>>(ws);
    absmax_kernel<<<1024, 256, 0, stream>>>(W, (unsigned long long)in_sizes[1], ws + 0);
    absmax_kernel<<<8,    256, 0, stream>>>(b, (unsigned long long)in_sizes[2], ws + 1);

    dim3 grid(OUT / TILE_N, M / TILE_M);

    const size_t xh_bytes = (size_t)M * IN * sizeof(_Float16);
    const size_t wq_bytes = (size_t)OUT * IN * sizeof(_Float16);
    const size_t need     = 16 + xh_bytes + wq_bytes;

    if (ws_size >= need) {
        // Pack to fp16 once; GEMM reads half the bytes and stages via async DMA.
        _Float16* xh = (_Float16*)((char*)d_ws + 16);
        _Float16* wq = (_Float16*)((char*)d_ws + 16 + xh_bytes);
        const unsigned long long nx4 = (unsigned long long)M * IN / 4;
        const unsigned long long nw4 = (unsigned long long)OUT * IN / 4;
        pack_x_kernel<<<(unsigned)((nx4 + 255) / 256), 256, 0, stream>>>(x, xh, nx4);
        pack_w_kernel<<<(unsigned)((nw4 + 255) / 256), 256, 0, stream>>>(
            W, wq, (const float*)d_ws, nw4);
        wmma_gemm_f16_async_kernel<<<grid, 256, 0, stream>>>(
            xh, wq, b, (const float*)d_ws, out, M, OUT, IN);
    } else {
        wmma_qlinear_fused_kernel<<<grid, 256, 0, stream>>>(
            x, W, b, (const float*)d_ws, out, M, OUT, IN);
    }
}
